// SwinTransformerBlock3D_63075889709439
// MI455X (gfx1250) — matrix operand order repaired and checked by reading, not compiled, not run on hardware
//
#include <hip/hip_runtime.h>
#include <hip/hip_bf16.h>

typedef __attribute__((ext_vector_type(16))) __bf16 v16bf;
typedef __attribute__((ext_vector_type(8)))  __bf16 v8bf;
typedef __attribute__((ext_vector_type(8)))  float  v8f;
typedef __attribute__((ext_vector_type(4)))  unsigned int u32x4;
typedef __attribute__((ext_vector_type(8)))  int i32x8;
typedef __attribute__((ext_vector_type(4)))  int i32x4;

#define NWIN   98
#define CDIM   384
#define NHEAD  12
#define MTOT   100352   // 1024 windows * 98 tokens

#if defined(__has_builtin)
#  if __has_builtin(__builtin_amdgcn_tensor_load_to_lds) && __has_builtin(__builtin_amdgcn_s_wait_tensorcnt)
#    define USE_TDM 1
#  else
#    define USE_TDM 0
#  endif
#else
#  define USE_TDM 0
#endif

static __device__ __forceinline__ void wmma_bf16(v8f& acc, const v16bf& a, const v16bf& b) {
  acc = __builtin_amdgcn_wmma_f32_16x16x32_bf16(false, a, false, b, (short)0, acc, false, false);
}

// ---------------------------------------------------------------------------
// fp32 -> bf16 elementwise convert (weights)
// ---------------------------------------------------------------------------
__global__ void cvt_bf16_kernel(const float* __restrict__ in, __bf16* __restrict__ out, int n) {
  int i = blockIdx.x * 256 + threadIdx.x;
  if (i < n) out[i] = (__bf16)in[i];
}

// ---------------------------------------------------------------------------
// Expand relative-position bias table -> bias[12][98][98]
// ---------------------------------------------------------------------------
__global__ void bias_expand_kernel(const float* __restrict__ bt, float* __restrict__ bM) {
  int i = blockIdx.x * 128 + threadIdx.x;
  if (i >= NHEAD * NWIN * NWIN) return;
  int head = i / (NWIN * NWIN);
  int rem  = i % (NWIN * NWIN);
  int n = rem / NWIN, m = rem % NWIN;
  int zdn = n / 49, r1 = n % 49, zhn = r1 / 7, zwn = r1 % 7;
  int zdm = m / 49, r2 = m % 49, zhm = r2 / 7, zwm = r2 % 7;
  int idx = (zdn - zdm + 1) * 169 + (zhn - zhm + 3) * 13 + (zwn - zwm + 3);
  bM[i] = bt[idx * NHEAD + head];
}

// ---------------------------------------------------------------------------
// LayerNorm (one wave per token row), MODE 0: fused shift-roll + window gather
// from x; MODE 1: identity gather from h. Output bf16 row-major [M][384].
// ---------------------------------------------------------------------------
template <int MODE>
__global__ void ln_kernel(const float* __restrict__ in, const float* __restrict__ g,
                          const float* __restrict__ bta, __bf16* __restrict__ out) {
  const int tid = threadIdx.x, lane = tid & 31, wave = tid >> 5;
  const size_t row = (size_t)blockIdx.x * 4 + wave;
  size_t src;
  if (MODE == 0) {
    int w = (int)(row / NWIN), n = (int)(row % NWIN);
    int b = w >> 9, rr = w & 511;
    int dI = rr >> 6, hI = (rr >> 3) & 7, wI = rr & 7;
    int wd = n / 49, r2 = n % 49, whh = r2 / 7, www = r2 % 7;
    int dp = dI * 2 + wd, hp = hI * 7 + whh, wp = wI * 7 + www;
    int sd = (dp + 1) & 15;
    int sh = hp + 3; if (sh >= 56) sh -= 56;
    int sw = wp + 3; if (sw >= 56) sw -= 56;
    src = ((size_t)b * 50176 + (size_t)sd * 3136 + (size_t)sh * 56 + sw) * CDIM;
  } else {
    src = row * CDIM;
  }
  float v[12];
  float s = 0.f, sq = 0.f;
#pragma unroll
  for (int i = 0; i < 12; ++i) {
    v[i] = in[src + lane + i * 32];
    s += v[i];
    sq += v[i] * v[i];
  }
#pragma unroll
  for (int off = 16; off > 0; off >>= 1) {
    s  += __shfl_xor(s, off, 32);
    sq += __shfl_xor(sq, off, 32);
  }
  float mean = s * (1.f / CDIM);
  float var  = sq * (1.f / CDIM) - mean * mean;
  float inv  = rsqrtf(var + 1e-5f);
  size_t drow = row * CDIM;
#pragma unroll
  for (int i = 0; i < 12; ++i) {
    int c = lane + i * 32;
    out[drow + c] = (__bf16)((v[i] - mean) * inv * g[c] + bta[c]);
  }
}

// ---------------------------------------------------------------------------
// Core WMMA GEMM mainloop: block tile 128(M) x 128(N), K-step 32, 8 waves
// (256 threads). Wave grid 4(M) x 2(N); wave tile 32 x 64 -> 8 wmma/K-step.
// A row-major [M][K] bf16 staged by the Tensor Data Mover (wave 0 issues,
// s_wait_tensorcnt + barrier), B row-major [K][N] bf16 transposed into LDS.
// No bounds checks: M%128==0, N%128==0, K%32==0 at all call sites.
// ---------------------------------------------------------------------------
static __device__ __forceinline__ void gemm_core(const __bf16* __restrict__ A,
                                                 const __bf16* __restrict__ B,
                                                 int K, int N, int M,
                                                 v8f acc[2][4]) {
  __shared__ __bf16 As[128 * 32];   // [m][k]
  __shared__ __bf16 Bt[128 * 32];   // [n][k]
  const int tid = threadIdx.x;
  const int lane = tid & 31, wave = tid >> 5;
  const int mA = lane & 15, hi = lane >> 4;
  const size_t bm = (size_t)blockIdx.y * 128;
  const int bn = blockIdx.x * 128;
  const int wm = (wave >> 1) * 32, wn = (wave & 1) * 64;
#if USE_TDM
  const unsigned lds_base = (unsigned)(unsigned long long)(void*)&As[0];
#endif
  for (int k0 = 0; k0 < K; k0 += 32) {
#if USE_TDM
    if (wave == 0) {
      // Tensor DMA: 2D tile (tile_dim0=32 elems along K, tile_dim1=128 rows)
      unsigned long long ga =
          (unsigned long long)(const void*)&A[bm * (size_t)K + k0];
      u32x4 g0;
      g0[0] = 1u;                                             // count=1
      g0[1] = lds_base;                                       // lds_addr
      g0[2] = (unsigned)ga;                                   // global addr lo
      g0[3] = (unsigned)((ga >> 32) & 0x01FFFFFFu) | 0x80000000u; // hi|type=2
      i32x8 g1;
      g1[0] = (int)(1u << 16);                                // data_size=2B
      g1[1] = (int)(((unsigned)K & 0xFFFFu) << 16);           // tensor_dim0 lo
      g1[2] = (int)((((unsigned)K >> 16) & 0xFFFFu) |
                    (((unsigned)M & 0xFFFFu) << 16));         // dim0 hi|dim1 lo
      g1[3] = (int)((((unsigned)M >> 16) & 0xFFFFu) |
                    (32u << 16));                             // dim1 hi|tile0=32
      g1[4] = (int)128u;                                      // tile1=128,tile2=0
      g1[5] = (int)(unsigned)K;                               // dim0_stride lo
      g1[6] = 0;                                              // stride hi
      g1[7] = 0;
      i32x4 z4 = {0, 0, 0, 0};
      i32x8 z8 = {0, 0, 0, 0, 0, 0, 0, 0};
      __builtin_amdgcn_tensor_load_to_lds(g0, g1, z4, z4, z8, 0);
    }
#else
#pragma unroll
    for (int i = 0; i < 2; ++i) {   // stage A: 128x32 bf16, 16B per thread-iter
      int idx = tid + i * 256;
      int r = idx >> 2, s = idx & 3;
      *(uint4*)(&As[r * 32 + s * 8]) =
          *(const uint4*)(&A[(bm + r) * (size_t)K + k0 + s * 8]);
    }
#endif
#pragma unroll
    for (int i = 0; i < 2; ++i) {   // stage B transposed: Bt[n][k]
      int idx = tid + i * 256;
      int kk = idx >> 4, s = idx & 15;
      v8bf vals = *(const v8bf*)(&B[(size_t)(k0 + kk) * N + bn + s * 8]);
#pragma unroll
      for (int j = 0; j < 8; ++j) Bt[(s * 8 + j) * 32 + kk] = vals[j];
    }
#if USE_TDM
    if (wave == 0) __builtin_amdgcn_s_wait_tensorcnt(0);
#endif
    __syncthreads();
    v16bf a[2];
#pragma unroll
    for (int mi = 0; mi < 2; ++mi) {
      const __bf16* ap = &As[(wm + mi * 16 + mA) * 32];
#pragma unroll
      for (int e = 0; e < 16; ++e)
        a[mi][e] = ap[(e & 7) + ((e >> 3) << 4) + hi * 8];
    }
#pragma unroll
    for (int nt = 0; nt < 4; ++nt) {
      v16bf b;
      const __bf16* bp = &Bt[(wn + nt * 16 + mA) * 32];
#pragma unroll
      for (int e = 0; e < 16; ++e) b[e] = bp[(e & 7) + ((e >> 3) << 4) + hi * 8];
      wmma_bf16(acc[0][nt], a[0], b);
      wmma_bf16(acc[1][nt], a[1], b);
    }
    __syncthreads();
  }
}

// ---------------------------------------------------------------------------
// GEMM 1: QKV. [M][384] x [384][1152] + bqkv. Scatter into per-(win,head)
// q/k/v buffers [win*12+head][98][32] bf16; q pre-scaled by hd^-0.5.
// ---------------------------------------------------------------------------
__global__ void gemm_qkv_kernel(const __bf16* __restrict__ A, const __bf16* __restrict__ B,
                                const float* __restrict__ bias,
                                __bf16* __restrict__ qout, __bf16* __restrict__ kout,
                                __bf16* __restrict__ vout) {
  v8f acc[2][4] = {};
  gemm_core(A, B, 384, 1152, MTOT, acc);
  const int lane = threadIdx.x & 31, wave = threadIdx.x >> 5;
  const int mA = lane & 15, hi = lane >> 4;
  const int wm = (wave >> 1) * 32, wn = (wave & 1) * 64;
  const size_t row00 = (size_t)blockIdx.y * 128 + wm + hi * 8;
  const int col0 = blockIdx.x * 128 + wn;
  const float scale = 0.17677669529663687f;   // 32^-0.5
  int rw[2][8], rn[2][8];
#pragma unroll
  for (int mi = 0; mi < 2; ++mi)
#pragma unroll
    for (int r = 0; r < 8; ++r) {
      size_t m = row00 + mi * 16 + r;
      rw[mi][r] = (int)(m / NWIN);
      rn[mi][r] = (int)(m % NWIN);
    }
#pragma unroll
  for (int nt = 0; nt < 4; ++nt) {
    int col = col0 + nt * 16 + mA;
    int t = col / CDIM, rem = col % CDIM;
    int hd = rem & 31, hh = rem >> 5;
    float bb = bias[col];
#pragma unroll
    for (int mi = 0; mi < 2; ++mi)
#pragma unroll
      for (int r = 0; r < 8; ++r) {
        size_t dst = ((size_t)(rw[mi][r] * NHEAD + hh) * NWIN + rn[mi][r]) * 32 + hd;
        float v = acc[mi][nt][r] + bb;
        if (t == 0)      qout[dst] = (__bf16)(v * scale);
        else if (t == 1) kout[dst] = (__bf16)v;
        else             vout[dst] = (__bf16)v;
      }
  }
}

// ---------------------------------------------------------------------------
// Attention: one block per (window,head). 98 padded to 112 (=7*16), K padded
// to 128 for the P@V GEMM. All matmuls via v_wmma_f32_16x16x32_bf16.
// ---------------------------------------------------------------------------
__global__ void attn_kernel(const __bf16* __restrict__ qb, const __bf16* __restrict__ kb,
                            const __bf16* __restrict__ vb, const float* __restrict__ biasM,
                            __bf16* __restrict__ ob) {
  __shared__ __bf16 q_s[112 * 32];     // [row][hd]
  __shared__ __bf16 k_s[112 * 32];     // [key][hd]
  __shared__ __bf16 vT_s[32 * 128];    // [hd][key] (transposed, K-padded)
  __shared__ float  s_s[112 * 112];    // scores
  __shared__ __bf16 p_s[112 * 128];    // probs, K-padded with zeros
  const int tid = threadIdx.x;
  const int lane = tid & 31, wave = tid >> 5;
  const int mA = lane & 15, hi = lane >> 4;
  const int wh = blockIdx.x;                 // w*12 + head
  const int head = wh % NHEAD, wnd = wh / NHEAD;
  const size_t base = (size_t)wh * NWIN * 32;

  // ---- stage q, k, v (v transposed), zero padding regions ----
  for (int i = tid; i < 112 * 32; i += 128) {
    __bf16 qv = (__bf16)0.0f, kv = (__bf16)0.0f;
    if (i < NWIN * 32) { qv = qb[base + i]; kv = kb[base + i]; }
    q_s[i] = qv;
    k_s[i] = kv;
  }
  for (int i = tid; i < NWIN * 32; i += 128) {
    int n = i >> 5, c = i & 31;
    vT_s[c * 128 + n] = vb[base + i];
  }
  for (int i = tid; i < 32 * 30; i += 128) {   // zero padded keys 98..127
    int c = i / 30, n = NWIN + i % 30;
    vT_s[c * 128 + n] = (__bf16)0.0f;
  }
  __syncthreads();

  // ---- S = q @ k^T + bias ----
  const float* bM = biasM + (size_t)head * NWIN * NWIN;
  for (int mt = wave; mt < 7; mt += 4) {
    v16bf a;
    const __bf16* ap = &q_s[(mt * 16 + mA) * 32];
#pragma unroll
    for (int e = 0; e < 16; ++e) a[e] = ap[(e & 7) + ((e >> 3) << 4) + hi * 8];
    for (int nt = 0; nt < 7; ++nt) {
      v16bf b;
      const __bf16* bp = &k_s[(nt * 16 + mA) * 32];
#pragma unroll
      for (int e = 0; e < 16; ++e) b[e] = bp[(e & 7) + ((e >> 3) << 4) + hi * 8];
      v8f acc = {};
      wmma_bf16(acc, a, b);
#pragma unroll
      for (int r = 0; r < 8; ++r) {
        int m = mt * 16 + r + hi * 8;
        int c = nt * 16 + mA;
        float sv = -1e30f;
        if (m < NWIN && c < NWIN) sv = acc[r] + bM[m * NWIN + c];
        s_s[m * 112 + c] = sv;
      }
    }
  }
  __syncthreads();

  // ---- row softmax, write bf16 probs with zero K-padding ----
  if (tid < 112) {
    int n = tid;
    if (n < NWIN) {
      float mx = -1e30f;
      for (int c = 0; c < NWIN; ++c) mx = fmaxf(mx, s_s[n * 112 + c]);
      float sum = 0.f;
      for (int c = 0; c < NWIN; ++c) sum += expf(s_s[n * 112 + c] - mx);
      float inv = 1.f / sum;
      for (int c = 0; c < NWIN; ++c)
        p_s[n * 128 + c] = (__bf16)(expf(s_s[n * 112 + c] - mx) * inv);
      for (int c = NWIN; c < 128; ++c) p_s[n * 128 + c] = (__bf16)0.0f;
    } else {
      for (int c = 0; c < 128; ++c) p_s[n * 128 + c] = (__bf16)0.0f;
    }
  }
  __syncthreads();

  // ---- O = P @ v  (M=112, K=128, N=32) ----
  for (int mt = wave; mt < 7; mt += 4) {
    v8f acc[2] = {};
    for (int kk = 0; kk < 4; ++kk) {
      v16bf a;
      const __bf16* ap = &p_s[(mt * 16 + mA) * 128 + kk * 32];
#pragma unroll
      for (int e = 0; e < 16; ++e) a[e] = ap[(e & 7) + ((e >> 3) << 4) + hi * 8];
#pragma unroll
      for (int nt = 0; nt < 2; ++nt) {
        v16bf b;
        const __bf16* bp = &vT_s[(nt * 16 + mA) * 128 + kk * 32];
#pragma unroll
        for (int e = 0; e < 16; ++e) b[e] = bp[(e & 7) + ((e >> 3) << 4) + hi * 8];
        wmma_bf16(acc[nt], a, b);
      }
    }
#pragma unroll
    for (int nt = 0; nt < 2; ++nt) {
#pragma unroll
      for (int r = 0; r < 8; ++r) {
        int m = mt * 16 + r + hi * 8;
        if (m < NWIN) {
          int c = head * 32 + nt * 16 + mA;
          ob[((size_t)wnd * NWIN + m) * CDIM + c] = (__bf16)acc[nt][r];
        }
      }
    }
  }
}

// ---------------------------------------------------------------------------
// GEMM 2: proj. Epilogue reverses window partition + roll and adds residual x.
// ---------------------------------------------------------------------------
__global__ void gemm_proj_kernel(const __bf16* __restrict__ A, const __bf16* __restrict__ B,
                                 const float* __restrict__ bias, const float* __restrict__ x,
                                 float* __restrict__ hout) {
  v8f acc[2][4] = {};
  gemm_core(A, B, 384, 384, MTOT, acc);
  const int lane = threadIdx.x & 31, wave = threadIdx.x >> 5;
  const int mA = lane & 15, hi = lane >> 4;
  const int wm = (wave >> 1) * 32, wn = (wave & 1) * 64;
  const size_t row00 = (size_t)blockIdx.y * 128 + wm + hi * 8;
  const int col0 = blockIdx.x * 128 + wn;
  size_t dib[2][8];
#pragma unroll
  for (int mi = 0; mi < 2; ++mi)
#pragma unroll
    for (int r = 0; r < 8; ++r) {
      size_t m = row00 + mi * 16 + r;
      int w = (int)(m / NWIN), n = (int)(m % NWIN);
      int b = w >> 9, rr = w & 511;
      int dI = rr >> 6, hI = (rr >> 3) & 7, wI = rr & 7;
      int wd = n / 49, r2 = n % 49, whh = r2 / 7, www = r2 % 7;
      int dp = dI * 2 + wd, hp = hI * 7 + whh, wp = wI * 7 + www;
      int sd = (dp + 1) & 15;
      int sh = hp + 3; if (sh >= 56) sh -= 56;
      int sw = wp + 3; if (sw >= 56) sw -= 56;
      dib[mi][r] = ((size_t)b * 50176 + (size_t)sd * 3136 + (size_t)sh * 56 + sw) * CDIM;
    }
#pragma unroll
  for (int nt = 0; nt < 4; ++nt) {
    int col = col0 + nt * 16 + mA;
    float bb = bias[col];
#pragma unroll
    for (int mi = 0; mi < 2; ++mi)
#pragma unroll
      for (int r = 0; r < 8; ++r) {
        size_t di = dib[mi][r] + col;
        hout[di] = x[di] + acc[mi][nt][r] + bb;
      }
  }
}

// ---------------------------------------------------------------------------
// GEMM 3: fc1 + exact GELU.
// ---------------------------------------------------------------------------
__global__ void gemm_fc1_kernel(const __bf16* __restrict__ A, const __bf16* __restrict__ B,
                                const float* __restrict__ bias, __bf16* __restrict__ out) {
  v8f acc[2][4] = {};
  gemm_core(A, B, 384, 1536, MTOT, acc);
  const int lane = threadIdx.x & 31, wave = threadIdx.x >> 5;
  const int mA = lane & 15, hi = lane >> 4;
  const int wm = (wave >> 1) * 32, wn = (wave & 1) * 64;
  const size_t row00 = (size_t)blockIdx.y * 128 + wm + hi * 8;
  const int col0 = blockIdx.x * 128 + wn;
#pragma unroll
  for (int nt = 0; nt < 4; ++nt) {
    int col = col0 + nt * 16 + mA;
    float bb = bias[col];
#pragma unroll
    for (int mi = 0; mi < 2; ++mi)
#pragma unroll
      for (int r = 0; r < 8; ++r) {
        float v = acc[mi][nt][r] + bb;
        float g = 0.5f * v * (1.f + erff(v * 0.70710678118654752f));
        out[(row00 + mi * 16 + r) * 1536 + col] = (__bf16)g;
      }
  }
}

// ---------------------------------------------------------------------------
// GEMM 4: fc2 + bias + residual h -> final output (fp32).
// ---------------------------------------------------------------------------
__global__ void gemm_fc2_kernel(const __bf16* __restrict__ A, const __bf16* __restrict__ B,
                                const float* __restrict__ bias, const float* __restrict__ hbuf,
                                float* __restrict__ out) {
  v8f acc[2][4] = {};
  gemm_core(A, B, 1536, 384, MTOT, acc);
  const int lane = threadIdx.x & 31, wave = threadIdx.x >> 5;
  const int mA = lane & 15, hi = lane >> 4;
  const int wm = (wave >> 1) * 32, wn = (wave & 1) * 64;
  const size_t row00 = (size_t)blockIdx.y * 128 + wm + hi * 8;
  const int col0 = blockIdx.x * 128 + wn;
#pragma unroll
  for (int nt = 0; nt < 4; ++nt) {
    int col = col0 + nt * 16 + mA;
    float bb = bias[col];
#pragma unroll
    for (int mi = 0; mi < 2; ++mi)
#pragma unroll
      for (int r = 0; r < 8; ++r) {
        size_t di = (row00 + mi * 16 + r) * CDIM + col;
        out[di] = hbuf[di] + acc[mi][nt][r] + bb;
      }
  }
}

// ---------------------------------------------------------------------------
extern "C" void kernel_launch(void* const* d_in, const int* in_sizes, int n_in,
                              void* d_out, int out_size, void* d_ws, size_t ws_size,
                              hipStream_t stream) {
  const float* x     = (const float*)d_in[0];
  const float* g1    = (const float*)d_in[4];
  const float* b1    = (const float*)d_in[5];
  const float* Wqkv  = (const float*)d_in[6];
  const float* bqkv  = (const float*)d_in[7];
  const float* btab  = (const float*)d_in[8];
  const float* Wproj = (const float*)d_in[9];
  const float* bproj = (const float*)d_in[10];
  const float* g2    = (const float*)d_in[11];
  const float* b2    = (const float*)d_in[12];
  const float* Wfc1  = (const float*)d_in[13];
  const float* bfc1  = (const float*)d_in[14];
  const float* Wfc2  = (const float*)d_in[15];
  const float* bfc2  = (const float*)d_in[16];
  float* outp = (float*)d_out;

  const size_t M = MTOT;
  char* ws = (char*)d_ws;
  size_t off = 0;
  auto alloc = [&](size_t bytes) -> char* {
    char* p = ws + off;
    off = (off + bytes + 255) & ~(size_t)255;
    return p;
  };
  float*  h_buf   = (float*)alloc(M * CDIM * 4);
  char*   region1 = alloc(M * 1536 * 2);       // hid; aliased earlier by xw + ob
  __bf16* hid     = (__bf16*)region1;
  __bf16* xw      = (__bf16*)region1;
  __bf16* ob      = (__bf16*)(region1 + M * CDIM * 2);
  char*   region2 = alloc(3 * M * CDIM * 2);   // q/k/v; aliased later by m_in
  __bf16* qb      = (__bf16*)region2;
  __bf16* kb      = (__bf16*)(region2 + M * CDIM * 2);
  __bf16* vb      = (__bf16*)(region2 + 2 * M * CDIM * 2);
  __bf16* m_in    = (__bf16*)region2;
  float*  biasM   = (float*)alloc((size_t)NHEAD * NWIN * NWIN * 4);
  __bf16* wqkv_bf = (__bf16*)alloc((size_t)384 * 1152 * 2);
  __bf16* wproj_bf= (__bf16*)alloc((size_t)384 * 384 * 2);
  __bf16* wfc1_bf = (__bf16*)alloc((size_t)384 * 1536 * 2);
  __bf16* wfc2_bf = (__bf16*)alloc((size_t)1536 * 384 * 2);

  // weight conversion + bias expansion
  cvt_bf16_kernel<<<(384 * 1152 + 255) / 256, 256, 0, stream>>>(Wqkv,  wqkv_bf,  384 * 1152);
  cvt_bf16_kernel<<<(384 * 384  + 255) / 256, 256, 0, stream>>>(Wproj, wproj_bf, 384 * 384);
  cvt_bf16_kernel<<<(384 * 1536 + 255) / 256, 256, 0, stream>>>(Wfc1,  wfc1_bf,  384 * 1536);
  cvt_bf16_kernel<<<(1536 * 384 + 255) / 256, 256, 0, stream>>>(Wfc2,  wfc2_bf,  1536 * 384);
  bias_expand_kernel<<<(NHEAD * NWIN * NWIN + 127) / 128, 128, 0, stream>>>(btab, biasM);

  // LN1 + shifted-window gather -> bf16 activations
  ln_kernel<0><<<M / 4, 128, 0, stream>>>(x, g1, b1, xw);

  // QKV GEMM (M=100352, K=384, N=1152)
  gemm_qkv_kernel<<<dim3(1152 / 128, M / 128), 256, 0, stream>>>(xw, wqkv_bf, bqkv, qb, kb, vb);

  // windowed attention (12288 window-head blocks)
  attn_kernel<<<1024 * NHEAD, 128, 0, stream>>>(qb, kb, vb, biasM, ob);

  // proj GEMM + window-reverse scatter + residual
  gemm_proj_kernel<<<dim3(CDIM / 128, M / 128), 256, 0, stream>>>(ob, wproj_bf, bproj, x, h_buf);

  // LN2 -> bf16
  ln_kernel<1><<<M / 4, 128, 0, stream>>>(h_buf, g2, b2, m_in);

  // MLP
  gemm_fc1_kernel<<<dim3(1536 / 128, M / 128), 256, 0, stream>>>(m_in, wfc1_bf, bfc1, hid);
  gemm_fc2_kernel<<<dim3(CDIM / 128, M / 128), 256, 0, stream>>>(hid, wfc2_bf, bfc2, h_buf, outp);

  (void)in_sizes; (void)n_in; (void)out_size; (void)ws_size;
}
